// LiGRUModel_19791209300526
// MI455X (gfx1250) — compile-verified, matrix operand order checked by reading, not run on hardware
//
#include <hip/hip_runtime.h>
#include <hip/hip_bf16.h>
#include <stdint.h>

// Problem constants
#define B_    64
#define T_    500
#define DIN   89
#define DINP  96      // pad 89 -> 96 (multiple of 32)
#define H_    512
#define G2    1024    // 2*H
#define OUTD  61
#define OUTP  64      // pad 61 -> 64

typedef __attribute__((ext_vector_type(16))) __bf16 v16bf;
typedef __attribute__((ext_vector_type(8)))  __bf16 v8bf;
typedef __attribute__((ext_vector_type(8)))  float  v8f;
typedef __attribute__((ext_vector_type(4)))  unsigned int v4u;
typedef __attribute__((ext_vector_type(8)))  int v8i;
typedef __attribute__((ext_vector_type(4)))  int v4i;

#define V8F_ZERO ((v8f){0.f,0.f,0.f,0.f,0.f,0.f,0.f,0.f})

#if defined(__has_builtin)
#if __has_builtin(__builtin_amdgcn_tensor_load_to_lds)
#define HAVE_TDM 1
#endif
#endif

// Load one 16x32 bf16 A/B fragment (row-major along K). Per CDNA5 ISA:
// lanes 0-15 hold row l, K = {0..7, 16..23}; lanes 16-31 hold row l, K = {8..15, 24..31}.
// `base` must already point at (row0, k0); ld in elements.
__device__ __forceinline__ v16bf load_frag(const __bf16* __restrict__ base, int ld, int lane) {
    int l  = lane & 15;
    int kh = (lane >> 4) << 3;               // 0 or 8
    const __bf16* p = base + (size_t)l * ld + kh;
    v8bf lo = *(const v8bf*)(p);             // K offsets +0..7   (16B)
    v8bf hi = *(const v8bf*)(p + 16);        // K offsets +16..23 (16B)
    return __builtin_shufflevector(lo, hi, 0,1,2,3,4,5,6,7,8,9,10,11,12,13,14,15);
}

__device__ __forceinline__ v8f wmma_bf16(v16bf a, v16bf b, v8f c) {
    // 8-arg form: (neg_a, A, neg_b, B, c_mod, C, reuse_a, reuse_b)
    return __builtin_amdgcn_wmma_f32_16x16x32_bf16(false, a, false, b, (short)0, c, false, false);
}

// Async global -> LDS copy of 16 bytes per lane (ASYNCcnt-tracked).
__device__ __forceinline__ void async_ld_b128(unsigned lds_off, const void* gaddr) {
    asm volatile("global_load_async_to_lds_b128 %0, %1, off"
                 :: "v"(lds_off), "v"((unsigned long long)(uintptr_t)gaddr)
                 : "memory");
}

// ---------------------------------------------------------------------------
// Small utility kernels
// ---------------------------------------------------------------------------
__global__ void k_zero_ctr(unsigned* c) { c[threadIdx.x] = 0u; }

// f32 [R][C] -> bf16 [Rp][Cp] with zero padding
__global__ void k_cvt_pad(const float* __restrict__ src, __bf16* __restrict__ dst,
                          int R, int C, int Rp, int Cp) {
    int i = blockIdx.x * 256 + threadIdx.x;
    int n = Rp * Cp;
    if (i >= n) return;
    int r = i / Cp, c = i % Cp;
    float v = (r < R && c < C) ? src[(size_t)r * C + c] : 0.0f;
    dst[i] = (__bf16)v;
}

// x [64][T][89] f32 -> X2 bf16 [T][128][96]; rows 64..127 are time-reversed copies
__global__ void k_pack_x0(const float* __restrict__ x, __bf16* __restrict__ X2) {
    int i = blockIdx.x * 256 + threadIdx.x;
    const int n = T_ * 128 * DINP;
    if (i >= n) return;
    int d   = i % DINP;
    int rt  = i / DINP;
    int row = rt % 128;
    int t   = rt / 128;
    int b   = row & 63;
    int tt  = (row < 64) ? t : (T_ - 1 - t);
    float v = (d < DIN) ? x[((size_t)b * T_ + tt) * DIN + d] : 0.0f;
    X2[i] = (__bf16)v;
}

// hs bf16 [T][128][512] -> X2 bf16 [T][128][1024] = concat(fwd, bwd) with time flips
__global__ void k_pack_next(const __bf16* __restrict__ hs, __bf16* __restrict__ X2) {
    int i = blockIdx.x * 256 + threadIdx.x;
    const int n = T_ * 128 * G2;             // 65,536,000 < 2^31
    if (i >= n) return;
    int c   = i % G2;
    int rt  = i / G2;
    int row = rt % 128;
    int t   = rt / 128;
    int b   = row & 63;
    __bf16 v;
    if (row < 64) {
        v = (c < H_) ? hs[((size_t)t * 128 + row) * H_ + c]
                     : hs[((size_t)(T_ - 1 - t) * 128 + 64 + b) * H_ + (c - H_)];
    } else {
        v = (c < H_) ? hs[((size_t)(T_ - 1 - t) * 128 + b) * H_ + c]
                     : hs[((size_t)t * 128 + row) * H_ + (c - H_)];
    }
    X2[i] = v;
}

// ---------------------------------------------------------------------------
// WMMA GEMM with async-copy double buffering:
// C_f32[M][N] = A_bf16[M][K](lda) @ B_bf16[N][K](ldb)^T
// Block = 256 thr (8 waves); block tile 64x128; wave tile 32x32.
// Per K-chunk the block stages A(64x32) + B(128x32) in LDS via
// global_load_async_to_lds_b128 (3 async ops per wave), double-buffered;
// WMMA consumes ds_load_b128 fragments while the next chunk is in flight.
// M % 64 == 0, N % 128 == 0, K % 32 == 0 (guaranteed by caller).
// ---------------------------------------------------------------------------
__device__ __forceinline__ void fill_tiles_async(
        const __bf16* __restrict__ Ablk, int lda,
        const __bf16* __restrict__ Bblk, int ldb, int k0,
        __bf16* sAbuf, __bf16* sBbuf, int tid) {
    // A tile: 64 rows x 32 cols = 256 chunks of 16B -> one per thread
    {
        int row = tid >> 2, qc = tid & 3;
        unsigned lds = (unsigned)(uintptr_t)(sAbuf + row * 32 + qc * 8);
        async_ld_b128(lds, Ablk + (size_t)row * lda + k0 + qc * 8);
    }
    // B tile: 128 rows x 32 cols = 512 chunks -> two per thread
    #pragma unroll
    for (int j = 0; j < 2; ++j) {
        int ch  = tid + j * 256;
        int row = ch >> 2, qc = ch & 3;
        unsigned lds = (unsigned)(uintptr_t)(sBbuf + row * 32 + qc * 8);
        async_ld_b128(lds, Bblk + (size_t)row * ldb + k0 + qc * 8);
    }
}

__global__ __launch_bounds__(256) void k_gemm_bf16(
        const __bf16* __restrict__ A, int lda,
        const __bf16* __restrict__ Bm, int ldb,
        float* __restrict__ C, int N, int K, int nTilesX) {
    __shared__ __bf16 sA[2][64 * 32];
    __shared__ __bf16 sB[2][128 * 32];

    int tid  = threadIdx.x;
    int lane = tid & 31;
    int wid  = tid >> 5;
    int bx   = blockIdx.x % nTilesX;
    int by   = blockIdx.x / nTilesX;
    int m0   = by * 64  + (wid >> 2) * 32;
    int n0   = bx * 128 + (wid & 3) * 32;
    int r0   = (wid >> 2) * 32;              // local A row base
    int c0l  = (wid & 3) * 32;               // local B row base

    const __bf16* Ablk = A  + (size_t)(by * 64) * lda;
    const __bf16* Bblk = Bm + (size_t)(bx * 128) * ldb;

    v8f acc00 = V8F_ZERO, acc01 = V8F_ZERO, acc10 = V8F_ZERO, acc11 = V8F_ZERO;

    fill_tiles_async(Ablk, lda, Bblk, ldb, 0, sA[0], sB[0], tid);

    int bu = 0;
    for (int k0 = 0; k0 < K; k0 += 32, bu ^= 1) {
        bool more = (k0 + 32 < K);
        if (more)
            fill_tiles_async(Ablk, lda, Bblk, ldb, k0 + 32, sA[bu ^ 1], sB[bu ^ 1], tid);
        // Async loads complete in order: waiting to <=3 outstanding drains the
        // current buffer's 3 ops while the next buffer's 3 stay in flight.
        if (more) asm volatile("s_wait_asynccnt 0x3" ::: "memory");
        else      asm volatile("s_wait_asynccnt 0x0" ::: "memory");
        __syncthreads();

        v16bf a0 = load_frag(&sA[bu][(size_t)r0 * 32], 32, lane);
        v16bf a1 = load_frag(&sA[bu][(size_t)(r0 + 16) * 32], 32, lane);
        v16bf b0 = load_frag(&sB[bu][(size_t)c0l * 32], 32, lane);
        v16bf b1 = load_frag(&sB[bu][(size_t)(c0l + 16) * 32], 32, lane);
        acc00 = wmma_bf16(a0, b0, acc00);
        acc01 = wmma_bf16(a0, b1, acc01);
        acc10 = wmma_bf16(a1, b0, acc10);
        acc11 = wmma_bf16(a1, b1, acc11);

        __syncthreads();                     // all reads done before buffer reuse
    }

    int hi8 = (lane >> 4) << 3;              // 0 or 8
    int nl  = lane & 15;
    #pragma unroll
    for (int r = 0; r < 8; ++r) {
        int ma = m0 + r + hi8, mb = ma + 16;
        int na = n0 + nl,      nb = na + 16;
        C[(size_t)ma * N + na] = acc00[r];
        C[(size_t)ma * N + nb] = acc01[r];
        C[(size_t)mb * N + na] = acc10[r];
        C[(size_t)mb * N + nb] = acc11[r];
    }
}

// ---------------------------------------------------------------------------
// Persistent LiGRU scan. grid = 16 blocks x 256 thr (128 waves).
// Block b works on one 16-row slice of h (m0 = (b/2)*16); its 8 waves cover 8
// 32-wide a-gate column strips (c0) plus paired z-gate strips (c0+512).
// Per step t: the block's h_{t-1} slice (16x512 bf16 = 16 KB) is staged into
// LDS by the Tensor Data Mover (one 2D descriptor issued by wave 0, waited
// with s_wait_tensorcnt), then g = w[t] + h_{t-1} @ U^T via WMMA,
// z = sigmoid(g_z), h = z*h + (1-z)*tanh(g_a); h kept in f32 regs.
// Steps are separated by a release/acquire atomic grid barrier.
// ---------------------------------------------------------------------------
__global__ __launch_bounds__(256) void k_ligru_scan(
        const float* __restrict__ w,       // [T][128][1024] f32
        const __bf16* __restrict__ U,      // [1024][512] bf16
        __bf16* __restrict__ hs,           // [T][128][512] bf16
        unsigned* __restrict__ ctr) {
    __shared__ __bf16 sH[16 * H_];         // 16 KB staged h_{t-1} slice

    int lane = threadIdx.x & 31;
    int wid  = threadIdx.x >> 5;
    int gw   = blockIdx.x * 8 + wid;       // 0..127
    int m0   = (gw >> 4) * 16;             // 0..112 (uniform per block)
    int c0   = (gw & 15) * 32;             // 0..480
    int hi8  = (lane >> 4) << 3;
    int nl   = lane & 15;

    const __bf16* Ub0 = U + (size_t)c0 * H_;
    const __bf16* Ub1 = U + (size_t)(c0 + 16) * H_;
    const __bf16* Ub2 = U + (size_t)(c0 + 512) * H_;
    const __bf16* Ub3 = U + (size_t)(c0 + 528) * H_;

    float hreg[16];
    #pragma unroll
    for (int i = 0; i < 16; ++i) hreg[i] = 0.0f;

    for (int t = 0; t < T_; ++t) {
        v8f ca0 = V8F_ZERO, ca1 = V8F_ZERO, cz0 = V8F_ZERO, cz1 = V8F_ZERO;
        if (t > 0) {
            const __bf16* hprev = hs + (size_t)(t - 1) * 128 * H_ + (size_t)m0 * H_;
#ifdef HAVE_TDM
            if (wid == 0) {                // wave-uniform branch: one TDM op per block
                unsigned lds_base = (unsigned)(uintptr_t)(&sH[0]);
                unsigned long long ga = (unsigned long long)(uintptr_t)hprev;
                v4u g0;
                g0[0] = 1u;                                        // count=1 (valid D#)
                g0[1] = lds_base;                                  // lds_addr
                g0[2] = (unsigned)(ga & 0xFFFFFFFFull);            // global_addr[31:0]
                g0[3] = (unsigned)((ga >> 32) & 0x1FFFFFFull) | (2u << 30); // addr[56:32] | type=2
                v8i g1;
                g1[0] = (int)(1u << 16);                           // data_size=1 (2 bytes)
                g1[1] = (int)((512u & 0xFFFFu) << 16);             // tensor_dim0 = 512
                g1[2] = (int)(16u << 16);                          // tensor_dim1 = 16
                g1[3] = (int)(512u << 16);                         // tile_dim0 = 512
                g1[4] = (int)16;                                   // tile_dim1 = 16, tile_dim2 = 0
                g1[5] = (int)512;                                  // tensor_dim0_stride = 512
                g1[6] = (int)((8192u & 0xFFFFu) << 16);            // tensor_dim1_stride = 8192 (lo16)
                g1[7] = 0;                                         // tensor_dim1_stride hi
                v4i g2 = {0, 0, 0, 0};
                v4i g3 = {0, 0, 0, 0};
                v8i g4 = {0, 0, 0, 0, 0, 0, 0, 0};                 // extra group (clang-23 6-arg form)
                __builtin_amdgcn_tensor_load_to_lds(g0, g1, g2, g3, g4, 0);
#if __has_builtin(__builtin_amdgcn_s_wait_tensorcnt)
                __builtin_amdgcn_s_wait_tensorcnt((short)0);
#else
                asm volatile("s_wait_tensorcnt 0x0" ::: "memory");
#endif
            }
#else
            // Fallback: cooperative 16B copies into LDS
            for (int i = threadIdx.x; i < (16 * H_) / 8; i += 256)
                ((v8bf*)sH)[i] = ((const v8bf*)hprev)[i];
#endif
            __syncthreads();
            for (int k0 = 0; k0 < H_; k0 += 32) {
                v16bf a  = load_frag(&sH[k0], H_, lane);
                v16bf b0 = load_frag(Ub0 + k0, H_, lane);
                v16bf b1 = load_frag(Ub1 + k0, H_, lane);
                v16bf b2 = load_frag(Ub2 + k0, H_, lane);
                v16bf b3 = load_frag(Ub3 + k0, H_, lane);
                ca0 = wmma_bf16(a, b0, ca0);
                ca1 = wmma_bf16(a, b1, ca1);
                cz0 = wmma_bf16(a, b2, cz0);
                cz1 = wmma_bf16(a, b3, cz1);
            }
        }
        const float* wt = w  + (size_t)t * 128 * G2;
        __bf16*      ht = hs + (size_t)t * 128 * H_;
        #pragma unroll
        for (int jt = 0; jt < 2; ++jt) {
            v8f ca = jt ? ca1 : ca0;
            v8f cz = jt ? cz1 : cz0;
            #pragma unroll
            for (int r = 0; r < 8; ++r) {
                int m  = m0 + r + hi8;
                int na = c0 + jt * 16 + nl;
                float a  = ca[r] + wt[(size_t)m * G2 + na];
                float zl = cz[r] + wt[(size_t)m * G2 + na + H_];
                float z  = 1.0f / (1.0f + __expf(-zl));
                float hn = z * hreg[jt * 8 + r] + (1.0f - z) * tanhf(a);
                hreg[jt * 8 + r] = hn;
                ht[(size_t)m * H_ + na] = (__bf16)hn;
            }
        }
        // Grid-wide barrier: hs[t] must be globally visible before step t+1.
        __threadfence();
        __syncthreads();
        if (threadIdx.x == 0) {
            __hip_atomic_fetch_add(ctr, 1u, __ATOMIC_RELEASE, __HIP_MEMORY_SCOPE_AGENT);
            unsigned target = (unsigned)(t + 1) * gridDim.x;
            while (__hip_atomic_load(ctr, __ATOMIC_ACQUIRE, __HIP_MEMORY_SCOPE_AGENT) < target) {
                __builtin_amdgcn_s_sleep(2);
            }
        }
        __syncthreads();
        __threadfence();
    }
}

// ---------------------------------------------------------------------------
// FC: out[b][t][o] = (sum_h X2[t][b][h] * fcw[o][h] + bias[o]) * (t < len[b])
// Rows r = t*64 + b.  Block = 8 waves x 32 rows = 256 rows; wave tile 32x64.
// ---------------------------------------------------------------------------
__global__ __launch_bounds__(256) void k_fc(
        const __bf16* __restrict__ X2,     // [T][128][1024]
        const __bf16* __restrict__ Wf,     // [64][1024]
        const float* __restrict__ bias,
        const int* __restrict__ len,
        float* __restrict__ out) {
    int lane = threadIdx.x & 31;
    int wid  = threadIdx.x >> 5;
    int r0   = blockIdx.x * 256 + wid * 32;      // row base; same t across the tile
    int t    = r0 / 64;
    int b0   = r0 % 64;
    const __bf16* A0 = X2 + ((size_t)t * 128 + b0) * G2;

    v8f acc[2][4];
    #pragma unroll
    for (int i = 0; i < 2; ++i)
        #pragma unroll
        for (int j = 0; j < 4; ++j) acc[i][j] = V8F_ZERO;

    for (int k0 = 0; k0 < G2; k0 += 32) {
        v16bf a0 = load_frag(A0 + k0, G2, lane);
        v16bf a1 = load_frag(A0 + (size_t)16 * G2 + k0, G2, lane);
        #pragma unroll
        for (int jt = 0; jt < 4; ++jt) {
            v16bf b = load_frag(Wf + (size_t)(jt * 16) * G2 + k0, G2, lane);
            acc[0][jt] = wmma_bf16(a0, b, acc[0][jt]);
            acc[1][jt] = wmma_bf16(a1, b, acc[1][jt]);
        }
    }

    int hi8 = (lane >> 4) << 3;
    int nl  = lane & 15;
    #pragma unroll
    for (int it = 0; it < 2; ++it) {
        #pragma unroll
        for (int jt = 0; jt < 4; ++jt) {
            #pragma unroll
            for (int r = 0; r < 8; ++r) {
                int row = r0 + it * 16 + r + hi8;
                int tt = row / 64, bb = row % 64, o = jt * 16 + nl;
                if (o < OUTD) {
                    float v = acc[it][jt][r] + bias[o];
                    out[((size_t)bb * T_ + tt) * OUTD + o] = (tt < len[bb]) ? v : 0.0f;
                }
            }
        }
    }
}

// ---------------------------------------------------------------------------
extern "C" void kernel_launch(void* const* d_in, const int* in_sizes, int n_in,
                              void* d_out, int out_size, void* d_ws, size_t ws_size,
                              hipStream_t stream) {
    const float* x    = (const float*)d_in[0];
    const int*   len  = (const int*)  d_in[1];
    const float* W0   = (const float*)d_in[2];
    const float* U0   = (const float*)d_in[3];
    const float* W1   = (const float*)d_in[4];
    const float* U1   = (const float*)d_in[5];
    const float* W2   = (const float*)d_in[6];
    const float* U2   = (const float*)d_in[7];
    const float* fcw  = (const float*)d_in[8];
    const float* fcb  = (const float*)d_in[9];
    float* out = (float*)d_out;

    char* p = (char*)d_ws;
    auto alloc = [&](size_t bytes) {
        char* r = p;
        p += (bytes + 255) & ~(size_t)255;
        return r;
    };
    float*  w    = (float*) alloc((size_t)T_ * 128 * G2 * sizeof(float));   // 262 MB
    __bf16* hs   = (__bf16*)alloc((size_t)T_ * 128 * H_ * sizeof(__bf16));  //  65 MB
    __bf16* X2   = (__bf16*)alloc((size_t)T_ * 128 * G2 * sizeof(__bf16));  // 131 MB
    __bf16* W0p  = (__bf16*)alloc((size_t)G2 * DINP * sizeof(__bf16));
    __bf16* W1p  = (__bf16*)alloc((size_t)G2 * G2 * sizeof(__bf16));
    __bf16* W2p  = (__bf16*)alloc((size_t)G2 * G2 * sizeof(__bf16));
    __bf16* U0p  = (__bf16*)alloc((size_t)G2 * H_ * sizeof(__bf16));
    __bf16* U1p  = (__bf16*)alloc((size_t)G2 * H_ * sizeof(__bf16));
    __bf16* U2p  = (__bf16*)alloc((size_t)G2 * H_ * sizeof(__bf16));
    __bf16* Wfp  = (__bf16*)alloc((size_t)OUTP * G2 * sizeof(__bf16));
    unsigned* ctr = (unsigned*)alloc(64 * sizeof(unsigned));

    k_zero_ctr<<<1, 64, 0, stream>>>(ctr);

    auto cvt = [&](const float* s, __bf16* d, int R, int C, int Rp, int Cp) {
        int n = Rp * Cp;
        k_cvt_pad<<<(n + 255) / 256, 256, 0, stream>>>(s, d, R, C, Rp, Cp);
    };
    cvt(W0,  W0p, G2,  DIN, G2,  DINP);
    cvt(W1,  W1p, G2,  G2,  G2,  G2);
    cvt(W2,  W2p, G2,  G2,  G2,  G2);
    cvt(U0,  U0p, G2,  H_,  G2,  H_);
    cvt(U1,  U1p, G2,  H_,  G2,  H_);
    cvt(U2,  U2p, G2,  H_,  G2,  H_);
    cvt(fcw, Wfp, OUTD, G2, OUTP, G2);

    k_pack_x0<<<(T_ * 128 * DINP + 255) / 256, 256, 0, stream>>>(x, X2);

    const int M = T_ * 128;                         // 64000 rows, % 64 == 0
    const int gemmBlocks = (M / 64) * (G2 / 128);   // 8000
    const int packBlocks = (T_ * 128 * G2 + 255) / 256;

    // ---- layer 0
    k_gemm_bf16<<<gemmBlocks, 256, 0, stream>>>(X2, DINP, W0p, DINP, w, G2, DINP, G2 / 128);
    k_ligru_scan<<<16, 256, 0, stream>>>(w, U0p, hs, ctr + 0);
    k_pack_next<<<packBlocks, 256, 0, stream>>>(hs, X2);
    // ---- layer 1
    k_gemm_bf16<<<gemmBlocks, 256, 0, stream>>>(X2, G2, W1p, G2, w, G2, G2, G2 / 128);
    k_ligru_scan<<<16, 256, 0, stream>>>(w, U1p, hs, ctr + 1);
    k_pack_next<<<packBlocks, 256, 0, stream>>>(hs, X2);
    // ---- layer 2
    k_gemm_bf16<<<gemmBlocks, 256, 0, stream>>>(X2, G2, W2p, G2, w, G2, G2, G2 / 128);
    k_ligru_scan<<<16, 256, 0, stream>>>(w, U2p, hs, ctr + 2);
    k_pack_next<<<packBlocks, 256, 0, stream>>>(hs, X2);

    // ---- FC + mask
    k_fc<<<(T_ * B_) / 256, 256, 0, stream>>>(X2, Wfp, fcb, len, out);

    (void)in_sizes; (void)n_in; (void)out_size; (void)ws_size;
}
